// CircularBoundaryBlock_46815143526781
// MI455X (gfx1250) — matrix operand order repaired and checked by reading, not compiled, not run on hardware
//
#include <hip/hip_runtime.h>
#include <stdint.h>

// ---------------------------------------------------------------------------
// CircularBoundaryBlock fused kernel for gfx1250 (MI455X)
//   prev/x/next gather -> GEMM1(384->128)+ReLU -> GEMM2(128->128)
//   -> +bias +residual -> LayerNorm -> ReLU
// GEMMs on v_wmma_f32_16x16x32_bf16 (f32 accumulate).
// Persistent workgroups: weights staged in LDS once per block; x tiles are
// double-buffered in LDS with the next tile's fill interleaved into the
// current tile's WMMA loop (prefetch hides HBM latency behind compute).
// h overwrites the current x buffer in place after a barrier.
// __launch_bounds__(256,2): cap VGPRs so 2 waves/SIMD stay resident (avoids
// >256-VGPR MSB addressing observed in round 4).
// ---------------------------------------------------------------------------

typedef __attribute__((ext_vector_type(16))) __bf16 v16bf;
typedef __attribute__((ext_vector_type(8)))  float  v8f;

union ABFrag { uint32_t u[8]; v16bf v; };

#define HDIM   128
#define NNODES 65536
#define NB     4

// geometry
#define THREADS 256
#define WAVES   8
#define ROWS_PER_WAVE 32                              // two 16-row strips
#define NODES_PER_TILE (WAVES * ROWS_PER_WAVE)        // 256
#define TILES_PER_BATCH (NNODES / NODES_PER_TILE)     // 256
#define NTILES (NB * TILES_PER_BATCH)                 // 1024
#define TILES_PER_BLOCK 4
#define GRID_MAIN (NTILES / TILES_PER_BLOCK)          // 256 persistent blocks

// packed-weight sizes (dwords): W1 12 kb x 8 nt x 8 v x 32 lanes, W2 4 kb x ..
#define W1P_DW (12 * 8 * 8 * 32)     // 24576
#define W2P_DW (4  * 8 * 8 * 32)     // 8192
#define WPK_DW (W1P_DW + W2P_DW)     // 32768 dwords = 128 KB

// LDS layout (bytes)
#define XB_STRIDE 132                                  // bf16 row stride (pad)
#define XROWS     (NODES_PER_TILE + 2)                 // 258 (circular halo)
#define XCHUNKS   (XROWS * 32)                         // 8256 float4 chunks
#define LDS_WPK_B   (WPK_DW * 4)                       // 131072
#define LDS_XB0_OFF LDS_WPK_B                          // 131072
#define LDS_XB_B    (XROWS * XB_STRIDE * 2)            // 68112
#define LDS_XB1_OFF (LDS_XB0_OFF + LDS_XB_B)           // 199184
#define SMEM_BYTES  (LDS_XB1_OFF + LDS_XB_B)           // 267296

__device__ __forceinline__ uint32_t pk_bf16(float a, float b) {
  uint32_t ua = __builtin_bit_cast(uint32_t, a);
  uint32_t ub = __builtin_bit_cast(uint32_t, b);
  ua += 0x7FFFu + ((ua >> 16) & 1u);          // round-to-nearest-even
  ub += 0x7FFFu + ((ub >> 16) & 1u);
  return (ua >> 16) | (ub & 0xFFFF0000u);
}

// ---------------------------------------------------------------------------
// Kernel 0: pack W1/W2 (f32, row-major [K x 128]) into bf16 B-fragment layout.
// Fragment dword index:  (((kb*8 + nt)*2 + j)*128 + L*4 + d), v = j*4 + d,
//   K = kb*32 + (L>=16 ? 16 : 0) + 2v (+1 in high half), N = nt*16 + (L&15).
// ---------------------------------------------------------------------------
__global__ __launch_bounds__(THREADS)
void pack_weights_kernel(const float* __restrict__ W1,
                         const float* __restrict__ W2,
                         uint32_t* __restrict__ wpk) {
  int i = blockIdx.x * blockDim.x + threadIdx.x;
  if (i < W1P_DW) {
    int Ld = i & 127, L = Ld >> 2, d = Ld & 3;
    int rest = i >> 7;
    int j  = rest & 1;
    int nt = (rest >> 1) & 7;
    int kb = rest >> 4;                       // 0..11
    int v  = j * 4 + d;
    int k  = kb * 32 + ((L & 16) ? 16 : 0) + 2 * v;
    int n  = nt * 16 + (L & 15);
    wpk[i] = pk_bf16(W1[k * HDIM + n], W1[(k + 1) * HDIM + n]);
  } else if (i < WPK_DW) {
    int i2 = i - W1P_DW;
    int Ld = i2 & 127, L = Ld >> 2, d = Ld & 3;
    int rest = i2 >> 7;
    int j  = rest & 1;
    int nt = (rest >> 1) & 7;
    int kb = rest >> 4;                       // 0..3
    int v  = j * 4 + d;
    int k  = kb * 32 + ((L & 16) ? 16 : 0) + 2 * v;
    int n  = nt * 16 + (L & 15);
    wpk[i] = pk_bf16(W2[k * HDIM + n], W2[(k + 1) * HDIM + n]);
  }
}

// load one float4 chunk of a tile and store packed bf16 pairs into LDS
__device__ __forceinline__ void fill_chunk(const float* __restrict__ x,
                                           size_t base, int nstart, int flat,
                                           __bf16* __restrict__ dst) {
  int r = flat >> 5, q = flat & 31;
  int n = (nstart - 1 + r) & (NNODES - 1);
  float4 v = *(const float4*)(x + base + (size_t)n * HDIM + q * 4);
  union { __bf16 h[4]; uint2 u; } cv;
  cv.h[0] = (__bf16)v.x; cv.h[1] = (__bf16)v.y;
  cv.h[2] = (__bf16)v.z; cv.h[3] = (__bf16)v.w;
  *(uint2*)&dst[r * XB_STRIDE + q * 4] = cv.u;
}

// ---------------------------------------------------------------------------
// Kernel 1: persistent fused block (4 tiles x 256 nodes per block)
// ---------------------------------------------------------------------------
__global__ __launch_bounds__(THREADS, 2)
void fused_persistent_kernel(const float* __restrict__ x,
                             const float* __restrict__ b1,
                             const float* __restrict__ b2,
                             const float* __restrict__ gamma,
                             const float* __restrict__ beta,
                             const uint32_t* __restrict__ wpkg,
                             float* __restrict__ out) {
  extern __shared__ char smem[];
  uint32_t* wpk = (uint32_t*)smem;                      // packed W1 | W2
  __bf16*   xb0 = (__bf16*)(smem + LDS_XB0_OFF);
  __bf16*   xb1 = (__bf16*)(smem + LDS_XB1_OFF);

  const int tid  = threadIdx.x;
  const int lane = tid & 31;
  const int wave = tid >> 5;

  const int m0   = wave * ROWS_PER_WAVE;
  const int mrow = lane & 15;
  const int hi8  = (lane & 16) ? 8 : 0;    // A K-offset hi lanes / D row+8
  const int nl   = lane & 15;

  // ---- once per block: weights -> LDS (coalesced, L2-hot) -----------------
  {
    const uint4* src = (const uint4*)wpkg;
    uint4*       dst = (uint4*)wpk;
    #pragma unroll
    for (int i = tid; i < WPK_DW / 4; i += THREADS) dst[i] = src[i];
  }
  // ---- once per block: fill first tile into xb0 ---------------------------
  {
    const int g0      = blockIdx.x * TILES_PER_BLOCK;
    const int batch0  = g0 >> 8;
    const int nstart0 = (g0 & (TILES_PER_BATCH - 1)) << 8;
    const size_t base0 = (size_t)batch0 * NNODES * HDIM;
    for (int flat = tid; flat < XCHUNKS; flat += THREADS)
      fill_chunk(x, base0, nstart0, flat, xb0);
  }
  __syncthreads();

  #pragma unroll 1
  for (int t = 0; t < TILES_PER_BLOCK; ++t) {
    const int g      = blockIdx.x * TILES_PER_BLOCK + t;
    const int batch  = g >> 8;
    const int nstart = (g & (TILES_PER_BATCH - 1)) << 8;
    __bf16* cur = (t & 1) ? xb1 : xb0;
    __bf16* nxt = (t & 1) ? xb0 : xb1;
    const bool has_next = (t + 1) < TILES_PER_BLOCK;
    const int gn       = g + 1;
    const int batchn   = gn >> 8;
    const int nstartn  = (gn & (TILES_PER_BATCH - 1)) << 8;
    const size_t basen = (size_t)batchn * NNODES * HDIM;

    // ===================== GEMM1: cat[32x384] @ W1 =========================
    v8f acc1[2][8] = {};
    for (int kb = 0; kb < 12; ++kb) {
      ABFrag a0, a1;
      #pragma unroll
      for (int v = 0; v < 8; ++v) {
        int k   = kb * 32 + (hi8 << 1) + (v << 1); // 0..383
        int seg = k >> 7;                  // prev/x/next
        int col = k & 127;
        int r   = m0 + mrow + seg;
        a0.u[v] = *(const uint32_t*)&cur[r * XB_STRIDE + col];
        a1.u[v] = *(const uint32_t*)&cur[(r + 16) * XB_STRIDE + col];
      }
      #pragma unroll
      for (int nt = 0; nt < 8; ++nt) {
        ABFrag b;
        int base = ((kb * 8 + nt) * 2) * 128 + lane * 4;
        *(uint4*)&b.u[0] = *(const uint4*)&wpk[base];
        *(uint4*)&b.u[4] = *(const uint4*)&wpk[base + 128];
        acc1[0][nt] = __builtin_amdgcn_wmma_f32_16x16x32_bf16(
            false, a0.v, false, b.v, (short)0, acc1[0][nt], false, false);
        acc1[1][nt] = __builtin_amdgcn_wmma_f32_16x16x32_bf16(
            false, a1.v, false, b.v, (short)0, acc1[1][nt], false, false);
      }
      // interleaved prefetch of next tile (3 chunks/thread/kstep covers all)
      if (has_next) {
        #pragma unroll
        for (int i = 0; i < 3; ++i) {
          int flat = (kb * 3 + i) * THREADS + tid;
          if (flat < XCHUNKS) fill_chunk(x, basen, nstartn, flat, nxt);
        }
      }
    }
    __syncthreads();   // all GEMM1 reads of cur done (halo rows shared)

    // ---- bias1 + ReLU; h (bf16, 32 rows) overwrites own rows of cur -------
    __bf16* hw = cur + m0 * XB_STRIDE;
    #pragma unroll
    for (int nt = 0; nt < 8; ++nt) {
      float bv = b1[nt * 16 + nl];
      #pragma unroll
      for (int s = 0; s < 2; ++s) {
        #pragma unroll
        for (int e = 0; e < 8; ++e) {
          float hval = fmaxf(acc1[s][nt][e] + bv, 0.0f);
          hw[(s * 16 + e + hi8) * XB_STRIDE + nt * 16 + nl] = (__bf16)hval;
        }
      }
    }
    // per-wave rows, LDS in-order within a wave: no barrier before GEMM2

    // ===================== GEMM2: h[32x128] @ W2 ===========================
    v8f acc2[2][8] = {};
    for (int kb = 0; kb < 4; ++kb) {
      ABFrag a0, a1;
      #pragma unroll
      for (int v = 0; v < 8; ++v) {
        int k = kb * 32 + (hi8 << 1) + (v << 1);   // 0..127
        a0.u[v] = *(const uint32_t*)&hw[mrow * XB_STRIDE + k];
        a1.u[v] = *(const uint32_t*)&hw[(16 + mrow) * XB_STRIDE + k];
      }
      #pragma unroll
      for (int nt = 0; nt < 8; ++nt) {
        ABFrag b;
        int base = W1P_DW + ((kb * 8 + nt) * 2) * 128 + lane * 4;
        *(uint4*)&b.u[0] = *(const uint4*)&wpk[base];
        *(uint4*)&b.u[4] = *(const uint4*)&wpk[base + 128];
        acc2[0][nt] = __builtin_amdgcn_wmma_f32_16x16x32_bf16(
            false, a0.v, false, b.v, (short)0, acc2[0][nt], false, false);
        acc2[1][nt] = __builtin_amdgcn_wmma_f32_16x16x32_bf16(
            false, a1.v, false, b.v, (short)0, acc2[1][nt], false, false);
      }
    }

    // ==== epilogue: +b2 +residual, LayerNorm (H=128), ReLU, store ==========
    float b2v[8], gv[8], btv[8];
    #pragma unroll
    for (int nt = 0; nt < 8; ++nt) {
      int c = nt * 16 + nl;
      b2v[nt] = b2[c]; gv[nt] = gamma[c]; btv[nt] = beta[c];
    }

    const size_t gbase = ((size_t)batch * NNODES + nstart + m0) * HDIM;
    #pragma unroll
    for (int s = 0; s < 2; ++s) {
      float psum[8] = {}, psq[8] = {};
      #pragma unroll
      for (int nt = 0; nt < 8; ++nt) {
        #pragma unroll
        for (int e = 0; e < 8; ++e) {
          // residual: re-read x (rows just streamed -> cache-hot, f32)
          float xr = x[gbase + (size_t)(s * 16 + e + hi8) * HDIM + nt * 16 + nl];
          float v  = acc2[s][nt][e] + b2v[nt] + xr;
          acc2[s][nt][e] = v;
          psum[e] += v;
          psq[e]  += v * v;
        }
      }
      #pragma unroll
      for (int e = 0; e < 8; ++e) {
        float s1 = psum[e], s2 = psq[e];
        // reduce across the 16 lanes holding this row (xor<=8 stays in half)
        #pragma unroll
        for (int mask = 8; mask >= 1; mask >>= 1) {
          s1 += __shfl_xor(s1, mask, 32);
          s2 += __shfl_xor(s2, mask, 32);
        }
        float mu   = s1 * (1.0f / HDIM);
        float var  = s2 * (1.0f / HDIM) - mu * mu;
        float rstd = rsqrtf(var + 1e-5f);
        float* orow = out + gbase + (size_t)(s * 16 + e + hi8) * HDIM;
        #pragma unroll
        for (int nt = 0; nt < 8; ++nt) {
          float y = (acc2[s][nt][e] - mu) * rstd * gv[nt] + btv[nt];
          orow[nt * 16 + nl] = fmaxf(y, 0.0f);
        }
      }
    }
    __syncthreads();   // cur free for next prefetch; nxt fully filled
  }
}

// ---------------------------------------------------------------------------
extern "C" void kernel_launch(void* const* d_in, const int* in_sizes, int n_in,
                              void* d_out, int out_size, void* d_ws, size_t ws_size,
                              hipStream_t stream) {
  const float* x     = (const float*)d_in[0];
  const float* W1    = (const float*)d_in[1];
  const float* b1    = (const float*)d_in[2];
  const float* W2    = (const float*)d_in[3];
  const float* b2    = (const float*)d_in[4];
  const float* gamma = (const float*)d_in[5];
  const float* beta  = (const float*)d_in[6];
  float*       out   = (float*)d_out;
  uint32_t*    wpk   = (uint32_t*)d_ws;   // 128 KB packed bf16 weights

  (void)hipFuncSetAttribute((const void*)fused_persistent_kernel,
                            hipFuncAttributeMaxDynamicSharedMemorySize,
                            (int)SMEM_BYTES);

  pack_weights_kernel<<<WPK_DW / THREADS, THREADS, 0, stream>>>(W1, W2, wpk);
  fused_persistent_kernel<<<GRID_MAIN, THREADS, SMEM_BYTES, stream>>>(
      x, b1, b2, gamma, beta, wpk, out);
}
